// LGCYMultiHeadAttention_1365799600500
// MI455X (gfx1250) — compile-verified, matrix operand order checked by reading, not run on hardware
//
#include <hip/hip_runtime.h>
#include <hip/hip_bf16.h>
#include <math.h>
#include <stdint.h>

// Problem constants (match reference)
#define D_MODEL 1024
#define DHEAD   64
#define NHEADS  16
#define BATCH   2
#define SEQ     2048
#define MTOT    (BATCH*SEQ)      // 4096 tokens
#define LN_EPS  1e-5f

// Tiling: block = 256 threads = 8 waves; block tile 256(M) x 64(N), K-chunk 64.
#define BLK_M 256
#define BLK_N 64
#define BLK_K 64
// LDS tiles are bf16, row = 64 data + 8 pad = 72 elts = 144B (16B-aligned
// fragment chunks, conflict-free lane gather). TDM reproduces this layout via
// pad_interval=32dw / pad_amount=4dw.
#define LDA   72
#define LDBT  72

typedef unsigned short bf16_t;  // storage type for kernel ABI (host-safe)
typedef __attribute__((ext_vector_type(16))) __bf16 v16bf;
typedef __attribute__((ext_vector_type(8)))  float  v8f;
typedef __attribute__((ext_vector_type(4)))  unsigned int v4u;
typedef __attribute__((ext_vector_type(4)))  unsigned int u32x4;
typedef __attribute__((ext_vector_type(8)))  int i32x8;
typedef __attribute__((ext_vector_type(4)))  int i32x4;

#define WAIT_ASYNC() asm volatile("s_wait_asynccnt 0x0" ::: "memory")

// ---------------------------------------------------------------------------
// TDM: one instruction DMAs a [rows x 64-bf16] tile (global row stride
// gstride elts) into padded LDS (144B rows) -- D# per CDNA5 ISA ch.8.
// Issue from a single wave; tracked by TENSORcnt. 6-arg builtin (clang-23).
// ---------------------------------------------------------------------------
__device__ __forceinline__ void tdm_load_tile(__bf16* sdst, const __bf16* gsrc,
                                              unsigned rows, unsigned gstride) {
  const uint64_t ga = (uint64_t)(uintptr_t)gsrc;
  u32x4 g0;
  g0[0] = 1u;                                        // count=1, user desc
  g0[1] = (unsigned)(uintptr_t)sdst;                 // lds_addr (low 32 bits)
  g0[2] = (unsigned)(ga & 0xffffffffu);              // global_addr[31:0]
  g0[3] = (unsigned)((ga >> 32) & 0x01ffffffu)       // global_addr[56:32]
          | (2u << 30);                              // type=2 (image)
  i32x8 g1;
  g1[0] = (int)((1u << 16)       // data_size = 2 bytes
                | (1u << 20)     // pad_enable
                | (4u << 22)     // pad_interval: 32 dwords (=128B row)
                | (3u << 25));   // pad_amount: 4 dwords (=16B pad)
  g1[1] = (int)((gstride & 0xffffu) << 16);          // tensor_dim0[15:0]
  g1[2] = (int)(((gstride >> 16) & 0xffffu)          // tensor_dim0[31:16]
                | ((rows & 0xffffu) << 16));         // tensor_dim1[15:0]
  g1[3] = (int)(((rows >> 16) & 0xffffu)             // tensor_dim1[31:16]
                | (64u << 16));                      // tile_dim0 = 64 elts
  g1[4] = (int)(rows & 0xffffu);                     // tile_dim1 = rows
  g1[5] = (int)gstride;                              // tensor_dim0_stride[31:0]
  g1[6] = 0;
  g1[7] = 0;
  i32x4 z4 = {0, 0, 0, 0};
  i32x8 z8 = {0, 0, 0, 0, 0, 0, 0, 0};
  __builtin_amdgcn_tensor_load_to_lds(g0, g1, z4, z4, z8, 0);
}

#define WAIT_TENSOR() __builtin_amdgcn_s_wait_tensorcnt(0)

// ---------------------------------------------------------------------------
// Per-lane async copy (ASYNCcnt) of a [ROWS x 64-bf16] tile into padded LDS.
// LDS byte address = low 32 bits of the generic pointer (ISA aperture rule).
// ---------------------------------------------------------------------------
template<int ROWS>
__device__ __forceinline__ void stage_async_64(__bf16* sdst, const __bf16* gsrc,
                                               int gstride, int tid) {
#pragma unroll
  for (int j = 0; j < (ROWS * 8) / 256; ++j) {
    int e = tid + j * 256;
    int r = e >> 3, c = e & 7;
    unsigned lds = (unsigned)(uintptr_t)(sdst + r * LDA) + (unsigned)(c * 16);
    const __bf16* g = gsrc + (size_t)r * gstride + c * 8;
    asm volatile("global_load_async_to_lds_b128 %0, %1, off"
                 :: "v"(lds), "v"(g) : "memory");
  }
}

// Fragment load: 16 contiguous bf16 (32B) from LDS as two b128 loads.
__device__ __forceinline__ v16bf frag_ld(const __bf16* p) {
  union { v16bf bf; v4u u[2]; } f;
  f.u[0] = *(const v4u*)p;
  f.u[1] = *(const v4u*)(p + 8);
  return f.bf;
}

__device__ __forceinline__ v8f wmma_bf16(v16bf a, v16bf b, v8f c) {
  return __builtin_amdgcn_wmma_f32_16x16x32_bf16(false, a, false, b, (short)0, c,
                                                 false, false);
}

// Convert float4 -> 4 bf16 and store 8 bytes (dst must be 8B aligned).
__device__ __forceinline__ void st_bf16x4(__bf16* dst, float4 v) {
  union { __bf16 h[4]; unsigned long long u64; } t;
  t.h[0] = (__bf16)v.x; t.h[1] = (__bf16)v.y;
  t.h[2] = (__bf16)v.z; t.h[3] = (__bf16)v.w;
  *(unsigned long long*)dst = t.u64;
}

// One 64-deep K-chunk: sA [BLK_M][LDA], sBT [BLK_N][LDBT] (bf16, k-contig).
__device__ __forceinline__ void mma_chunk(const __bf16* sA, const __bf16* sBT,
                                          int wave, int lane, v8f acc[8]) {
  const int lrow = lane & 15;
  const int khalf = (lane >> 4) << 4;  // 0 or 16
#pragma unroll
  for (int ks = 0; ks < 2; ++ks) {
    const int k0 = ks * 32 + khalf;
    v16bf a0 = frag_ld(sA + (wave * 32 + lrow) * LDA + k0);
    v16bf a1 = frag_ld(sA + (wave * 32 + 16 + lrow) * LDA + k0);
#pragma unroll
    for (int nt = 0; nt < 4; ++nt) {
      v16bf b = frag_ld(sBT + (nt * 16 + lrow) * LDBT + k0);
      acc[nt]     = wmma_bf16(a0, b, acc[nt]);
      acc[4 + nt] = wmma_bf16(a1, b, acc[4 + nt]);
    }
  }
}

// ---------------------------------------------------------------------------
// Pre-pass A: f32 -> bf16 elementwise (q, k, v inputs).
// ---------------------------------------------------------------------------
__global__ __launch_bounds__(256)
void cvt_bf16_kernel(const float* __restrict__ src, bf16_t* __restrict__ dstu,
                     int n4) {
  __bf16* dst = (__bf16*)dstu;
  int i = blockIdx.x * 256 + threadIdx.x;
  if (i < n4) {
    float4 v = ((const float4*)src)[i];
    st_bf16x4(dst + (size_t)i * 4, v);
  }
}

// ---------------------------------------------------------------------------
// Pre-pass B: weight [1024k][1024n] f32 -> transposed bf16 Wt[n][k].
// ---------------------------------------------------------------------------
__global__ __launch_bounds__(256)
void transpose_cvt_kernel(const float* __restrict__ W, bf16_t* __restrict__ Wtu) {
  __bf16* Wt = (__bf16*)Wtu;
  __shared__ float t[32][33];
  const int bn = blockIdx.x * 32, bk = blockIdx.y * 32;
  const int x = threadIdx.x & 31, y = threadIdx.x >> 5;  // 32 x 8
#pragma unroll
  for (int j = y; j < 32; j += 8)
    t[j][x] = W[(size_t)(bk + j) * D_MODEL + bn + x];
  __syncthreads();
#pragma unroll
  for (int j = y; j < 32; j += 8)
    Wt[(size_t)(bn + j) * D_MODEL + bk + x] = (__bf16)t[x][j];
}

// ---------------------------------------------------------------------------
// Kernel 1: projection GEMM (bf16 in / bf16 out). A tile via TDM (wave 0),
// B tile via per-lane async copies; double-buffered, overlapped with WMMA.
// MODE 0: Out[b,h,s,c] (Q,K).  MODE 1: Out[b,h,c,s] (V transposed for PV GEMM).
// ---------------------------------------------------------------------------
template<int MODE>
__global__ __launch_bounds__(256)
void proj_kernel(const bf16_t* __restrict__ Xu, const bf16_t* __restrict__ Wtu,
                 const float* __restrict__ bias, bf16_t* __restrict__ Outu) {
  const __bf16* Xb = (const __bf16*)Xu;
  const __bf16* Wt = (const __bf16*)Wtu;
  __bf16* Out = (__bf16*)Outu;

  __shared__ __bf16 sA[2 * BLK_M * LDA];
  __shared__ __bf16 sB[2 * BLK_N * LDBT];

  const int tid = threadIdx.x, lane = tid & 31, wave = tid >> 5;
  const int m0 = blockIdx.x * BLK_M;
  const int n0 = blockIdx.y * BLK_N;

  v8f acc[8] = {};

  if (tid < 32)
    tdm_load_tile(sA, Xb + (size_t)m0 * D_MODEL, BLK_M, D_MODEL);
  stage_async_64<BLK_N>(sB, Wt + (size_t)n0 * D_MODEL, D_MODEL, tid);

  const int NIT = D_MODEL / BLK_K;  // 16
  for (int it = 0; it < NIT; ++it) {
    const int cur = it & 1;
    WAIT_ASYNC();
    if (tid < 32) WAIT_TENSOR();
    __syncthreads();
    if (it + 1 < NIT) {
      const int kc = (it + 1) * BLK_K;
      if (tid < 32)
        tdm_load_tile(sA + (cur ^ 1) * BLK_M * LDA,
                      Xb + (size_t)m0 * D_MODEL + kc, BLK_M, D_MODEL);
      stage_async_64<BLK_N>(sB + (cur ^ 1) * BLK_N * LDBT,
                            Wt + (size_t)n0 * D_MODEL + kc, D_MODEL, tid);
    }
    mma_chunk(sA + cur * BLK_M * LDA, sB + cur * BLK_N * LDBT, wave, lane, acc);
    __syncthreads();
  }

#pragma unroll
  for (int nt = 0; nt < 4; ++nt) {
    int n = n0 + nt * 16 + (lane & 15);
    int h = n >> 6, c = n & 63;
    float bv = bias[n];
#pragma unroll
    for (int ms = 0; ms < 2; ++ms) {
#pragma unroll
      for (int vv = 0; vv < 8; ++vv) {
        int m = m0 + wave * 32 + ms * 16 + ((lane < 16) ? vv : vv + 8);
        int b = m >> 11, s = m & (SEQ - 1);
        __bf16 r = (__bf16)(acc[ms * 4 + nt][vv] + bv);
        if (MODE == 0)
          Out[(((size_t)(b * NHEADS + h) * SEQ) + s) * DHEAD + c] = r;
        else
          Out[(((size_t)(b * NHEADS + h) * DHEAD) + c) * SEQ + s] = r;
      }
    }
  }
}

// ---------------------------------------------------------------------------
// Kernel 2: scores = scale * Q K^T, masked. Q tile via TDM, K tile via async.
// ---------------------------------------------------------------------------
__global__ __launch_bounds__(256)
void scores_kernel(const bf16_t* __restrict__ Qu, const bf16_t* __restrict__ Ku,
                   const unsigned char* __restrict__ mask,
                   float* __restrict__ attn) {
  const __bf16* Qp = (const __bf16*)Qu;
  const __bf16* Kp = (const __bf16*)Ku;
  __shared__ __bf16 sA[BLK_M * LDA];
  __shared__ __bf16 sB[BLK_N * LDBT];

  const int tid = threadIdx.x, lane = tid & 31, wave = tid >> 5;
  const int q0 = blockIdx.x * BLK_M;
  const int k0 = blockIdx.y * BLK_N;
  const int bh = blockIdx.z;
  const int bb = bh >> 4;
  const __bf16* Qb = Qp + (size_t)bh * SEQ * DHEAD;
  const __bf16* Kb = Kp + (size_t)bh * SEQ * DHEAD;

  v8f acc[8] = {};

  if (tid < 32)
    tdm_load_tile(sA, Qb + (size_t)q0 * DHEAD, BLK_M, DHEAD);
  stage_async_64<BLK_N>(sB, Kb + (size_t)k0 * DHEAD, DHEAD, tid);
  WAIT_ASYNC();
  if (tid < 32) WAIT_TENSOR();
  __syncthreads();
  mma_chunk(sA, sB, wave, lane, acc);

  const float scale = 0.125f;  // 1/sqrt(64)
#pragma unroll
  for (int nt = 0; nt < 4; ++nt) {
    int kk = k0 + nt * 16 + (lane & 15);
#pragma unroll
    for (int ms = 0; ms < 2; ++ms) {
#pragma unroll
      for (int vv = 0; vv < 8; ++vv) {
        int qq = q0 + wave * 32 + ms * 16 + ((lane < 16) ? vv : vv + 8);
        float val = acc[ms * 4 + nt][vv] * scale;
        if (mask[((size_t)bb * SEQ + qq) * SEQ + kk]) val = 1e9f;
        attn[((size_t)bh * SEQ + qq) * SEQ + kk] = val;
      }
    }
  }
}

// ---------------------------------------------------------------------------
// Kernel 3: row softmax over 2048, in place. One block per row.
// ---------------------------------------------------------------------------
__global__ __launch_bounds__(256)
void softmax_kernel(float* __restrict__ attn) {
  const size_t row = blockIdx.x;
  float* p = attn + row * SEQ;
  const int tid = threadIdx.x;
  __shared__ float red[8];

  float vals[8];
  float m = -3.4e38f;
#pragma unroll
  for (int j = 0; j < 8; ++j) {
    vals[j] = p[tid + j * 256];
    m = fmaxf(m, vals[j]);
  }
#pragma unroll
  for (int off = 16; off > 0; off >>= 1) m = fmaxf(m, __shfl_xor(m, off, 32));
  if ((tid & 31) == 0) red[tid >> 5] = m;
  __syncthreads();
  float bm = red[0];
#pragma unroll
  for (int i = 1; i < 8; ++i) bm = fmaxf(bm, red[i]);
  __syncthreads();

  float s = 0.f;
#pragma unroll
  for (int j = 0; j < 8; ++j) {
    vals[j] = __expf(vals[j] - bm);
    s += vals[j];
  }
#pragma unroll
  for (int off = 16; off > 0; off >>= 1) s += __shfl_xor(s, off, 32);
  if ((tid & 31) == 0) red[tid >> 5] = s;
  __syncthreads();
  float bs = 0.f;
#pragma unroll
  for (int i = 0; i < 8; ++i) bs += red[i];
  float inv = 1.0f / bs;
#pragma unroll
  for (int j = 0; j < 8; ++j) p[tid + j * 256] = vals[j] * inv;
}

// ---------------------------------------------------------------------------
// Kernel 4: context = P @ V per (b,h). A = f32 probs (cvt-staged),
// B = pre-transposed bf16 V (async-staged). ctx out: [B,S,1024] bf16.
// ---------------------------------------------------------------------------
__global__ __launch_bounds__(256)
void context_kernel(const float* __restrict__ attn, const bf16_t* __restrict__ Vtu,
                    bf16_t* __restrict__ ctxu) {
  const __bf16* Vt = (const __bf16*)Vtu;
  __bf16* ctx = (__bf16*)ctxu;
  __shared__ __bf16 sA[BLK_M * LDA];
  __shared__ __bf16 sB[BLK_N * LDBT];

  const int tid = threadIdx.x, lane = tid & 31, wave = tid >> 5;
  const int q0 = blockIdx.x * BLK_M;
  const int bh = blockIdx.y;
  const float* Pb = attn + (size_t)bh * SEQ * SEQ;
  const __bf16* Vb = Vt + (size_t)bh * DHEAD * SEQ;  // [c][s]

  v8f acc[8] = {};

  for (int kc = 0; kc < SEQ; kc += BLK_K) {
    // B: async copy of V^T tile [64c][64k]
    stage_async_64<BLK_N>(sB, Vb + kc, SEQ, tid);
    // A: cvt-stage P tile [256][64] (f32 -> bf16)
#pragma unroll
    for (int j = 0; j < 16; ++j) {
      int e = tid + j * 256;
      int r = e >> 4, c4 = e & 15;
      float4 v = *(const float4*)&Pb[(size_t)(q0 + r) * SEQ + kc + c4 * 4];
      st_bf16x4(&sA[r * LDA + c4 * 4], v);
    }
    WAIT_ASYNC();
    __syncthreads();
    mma_chunk(sA, sB, wave, lane, acc);
    __syncthreads();
  }

  const int bb = bh >> 4, h = bh & 15;
#pragma unroll
  for (int nt = 0; nt < 4; ++nt) {
    int c = nt * 16 + (lane & 15);
#pragma unroll
    for (int ms = 0; ms < 2; ++ms) {
#pragma unroll
      for (int vv = 0; vv < 8; ++vv) {
        int s = q0 + wave * 32 + ms * 16 + ((lane < 16) ? vv : vv + 8);
        ctx[((size_t)(bb * SEQ + s)) * D_MODEL + h * DHEAD + c] =
            (__bf16)acc[ms * 4 + nt][vv];
      }
    }
  }
}

// ---------------------------------------------------------------------------
// Kernel 5: out-proj GEMM (TDM A tile + async B tile) + bias + residual(q).
// ---------------------------------------------------------------------------
__global__ __launch_bounds__(256)
void outproj_kernel(const bf16_t* __restrict__ ctxu, const bf16_t* __restrict__ Wotu,
                    const float* __restrict__ bo, const float* __restrict__ qin,
                    float* __restrict__ xbuf) {
  const __bf16* Xb = (const __bf16*)ctxu;
  const __bf16* Wt = (const __bf16*)Wotu;
  __shared__ __bf16 sA[2 * BLK_M * LDA];
  __shared__ __bf16 sB[2 * BLK_N * LDBT];

  const int tid = threadIdx.x, lane = tid & 31, wave = tid >> 5;
  const int m0 = blockIdx.x * BLK_M;
  const int n0 = blockIdx.y * BLK_N;

  v8f acc[8] = {};

  if (tid < 32)
    tdm_load_tile(sA, Xb + (size_t)m0 * D_MODEL, BLK_M, D_MODEL);
  stage_async_64<BLK_N>(sB, Wt + (size_t)n0 * D_MODEL, D_MODEL, tid);

  const int NIT = D_MODEL / BLK_K;
  for (int it = 0; it < NIT; ++it) {
    const int cur = it & 1;
    WAIT_ASYNC();
    if (tid < 32) WAIT_TENSOR();
    __syncthreads();
    if (it + 1 < NIT) {
      const int kc = (it + 1) * BLK_K;
      if (tid < 32)
        tdm_load_tile(sA + (cur ^ 1) * BLK_M * LDA,
                      Xb + (size_t)m0 * D_MODEL + kc, BLK_M, D_MODEL);
      stage_async_64<BLK_N>(sB + (cur ^ 1) * BLK_N * LDBT,
                            Wt + (size_t)n0 * D_MODEL + kc, D_MODEL, tid);
    }
    mma_chunk(sA + cur * BLK_M * LDA, sB + cur * BLK_N * LDBT, wave, lane, acc);
    __syncthreads();
  }

#pragma unroll
  for (int nt = 0; nt < 4; ++nt) {
    int n = n0 + nt * 16 + (lane & 15);
    float bv = bo[n];
#pragma unroll
    for (int ms = 0; ms < 2; ++ms) {
#pragma unroll
      for (int vv = 0; vv < 8; ++vv) {
        int m = m0 + wave * 32 + ms * 16 + ((lane < 16) ? vv : vv + 8);
        size_t idx = (size_t)m * D_MODEL + n;
        xbuf[idx] = acc[ms * 4 + nt][vv] + bv + qin[idx];
      }
    }
  }
}

// ---------------------------------------------------------------------------
// Kernel 6: layernorm per row of 1024 -> final output
// ---------------------------------------------------------------------------
__global__ __launch_bounds__(256)
void layernorm_kernel(const float* __restrict__ x, const float* __restrict__ g,
                      const float* __restrict__ beta, float* __restrict__ out) {
  const size_t row = blockIdx.x;
  const float* p = x + row * D_MODEL;
  const int tid = threadIdx.x;
  __shared__ float red[8];

  float v[4];
  float s = 0.f;
#pragma unroll
  for (int j = 0; j < 4; ++j) {
    v[j] = p[tid + j * 256];
    s += v[j];
  }
#pragma unroll
  for (int off = 16; off > 0; off >>= 1) s += __shfl_xor(s, off, 32);
  if ((tid & 31) == 0) red[tid >> 5] = s;
  __syncthreads();
  float tot = 0.f;
#pragma unroll
  for (int i = 0; i < 8; ++i) tot += red[i];
  const float mu = tot * (1.0f / D_MODEL);
  __syncthreads();

  float d2 = 0.f;
#pragma unroll
  for (int j = 0; j < 4; ++j) {
    float d = v[j] - mu;
    d2 += d * d;
  }
#pragma unroll
  for (int off = 16; off > 0; off >>= 1) d2 += __shfl_xor(d2, off, 32);
  if ((tid & 31) == 0) red[tid >> 5] = d2;
  __syncthreads();
  float vtot = 0.f;
#pragma unroll
  for (int i = 0; i < 8; ++i) vtot += red[i];
  const float inv = rsqrtf(vtot * (1.0f / D_MODEL) + LN_EPS);

#pragma unroll
  for (int j = 0; j < 4; ++j) {
    int c = tid + j * 256;
    out[row * D_MODEL + c] = (v[j] - mu) * inv * g[c] + beta[c];
  }
}

// ---------------------------------------------------------------------------
extern "C" void kernel_launch(void* const* d_in, const int* in_sizes, int n_in,
                              void* d_out, int out_size, void* d_ws, size_t ws_size,
                              hipStream_t stream) {
  const float* q  = (const float*)d_in[0];
  const float* k  = (const float*)d_in[1];
  const float* v  = (const float*)d_in[2];
  const unsigned char* mask = (const unsigned char*)d_in[3];
  const float* Wq = (const float*)d_in[4];
  const float* bq = (const float*)d_in[5];
  const float* Wk = (const float*)d_in[6];
  const float* bk = (const float*)d_in[7];
  const float* Wv = (const float*)d_in[8];
  const float* bv = (const float*)d_in[9];
  const float* Wo = (const float*)d_in[10];
  const float* bo = (const float*)d_in[11];
  const float* ln_g = (const float*)d_in[12];
  const float* ln_b = (const float*)d_in[13];

  float* out  = (float*)d_out;                       // [B,S,1024] f32
  float* attn = out + (size_t)MTOT * D_MODEL;        // [B,H,S,S] f32

  // workspace layout (byte offsets)
  char* w = (char*)d_ws;
  bf16_t* qb  = (bf16_t*)(w);                        // 8MB  [4096,1024] bf16
  bf16_t* kb  = (bf16_t*)(w + (8 << 20));
  bf16_t* vb  = (bf16_t*)(w + (16 << 20));
  bf16_t* Wqt = (bf16_t*)(w + (24 << 20));           // 2MB each, transposed
  bf16_t* Wkt = (bf16_t*)(w + (26 << 20));
  bf16_t* Wvt = (bf16_t*)(w + (28 << 20));
  bf16_t* Wot = (bf16_t*)(w + (30 << 20));
  bf16_t* Qb  = (bf16_t*)(w + (32 << 20));           // 8MB [B,H,S,64]
  bf16_t* Kb  = (bf16_t*)(w + (40 << 20));           // 8MB [B,H,S,64]
  bf16_t* Vt  = (bf16_t*)(w + (48 << 20));           // 8MB [B,H,64,S]
  bf16_t* ctxb= (bf16_t*)(w + (56 << 20));           // 8MB [B,S,1024]
  float*  xb  = (float*) (w + (64 << 20));           // 16MB [4096,1024] f32

  dim3 blk(256);

  // pre-passes: bf16 conversions / weight transposes
  const int n4 = MTOT * D_MODEL / 4;
  cvt_bf16_kernel<<<dim3(n4 / 256), blk, 0, stream>>>(q, qb, n4);
  cvt_bf16_kernel<<<dim3(n4 / 256), blk, 0, stream>>>(k, kb, n4);
  cvt_bf16_kernel<<<dim3(n4 / 256), blk, 0, stream>>>(v, vb, n4);
  dim3 gT(D_MODEL / 32, D_MODEL / 32);
  transpose_cvt_kernel<<<gT, blk, 0, stream>>>(Wq, Wqt);
  transpose_cvt_kernel<<<gT, blk, 0, stream>>>(Wk, Wkt);
  transpose_cvt_kernel<<<gT, blk, 0, stream>>>(Wv, Wvt);
  transpose_cvt_kernel<<<gT, blk, 0, stream>>>(Wo, Wot);

  dim3 gProj(MTOT / BLK_M, D_MODEL / BLK_N);         // 16 x 16
  proj_kernel<0><<<gProj, blk, 0, stream>>>(qb, Wqt, bq, Qb);
  proj_kernel<0><<<gProj, blk, 0, stream>>>(kb, Wkt, bk, Kb);
  proj_kernel<1><<<gProj, blk, 0, stream>>>(vb, Wvt, bv, Vt);

  dim3 gS(SEQ / BLK_M, SEQ / BLK_N, BATCH * NHEADS); // 8 x 32 x 32
  scores_kernel<<<gS, blk, 0, stream>>>(Qb, Kb, mask, attn);

  softmax_kernel<<<dim3(BATCH * NHEADS * SEQ), blk, 0, stream>>>(attn);

  dim3 gC(SEQ / BLK_M, BATCH * NHEADS);              // 8 x 32
  context_kernel<<<gC, blk, 0, stream>>>(attn, Vt, ctxb);

  outproj_kernel<<<gProj, blk, 0, stream>>>(ctxb, Wot, bo, q, xb);

  layernorm_kernel<<<dim3(MTOT), blk, 0, stream>>>(xb, ln_g, ln_b, out);
}
